// DeepPMSeq_76776835383837
// MI455X (gfx1250) — matrix operand order repaired and checked by reading, not compiled, not run on hardware
//
#include <hip/hip_runtime.h>

typedef __attribute__((ext_vector_type(16))) _Float16 v16h;
typedef __attribute__((ext_vector_type(8)))  _Float16 v8h;
typedef __attribute__((ext_vector_type(8)))  float    v8f;

#define L_    4
#define D_    256
#define H_    8
#define HD_   32
#define DFF_  1024
#define S_    64
#define NSEQ  1024

// LDS layout (bytes)
#define XF_OFF   0        // 64x256 f16 activation (32768 B)
#define QB_OFF   32768    // Q f16, later reused as attention-output O (32768 B)
#define KB_OFF   65536    // K f16 (32768 B)
#define VT_OFF   98304    // V^T f16 [256][64] (32768 B)
#define ATT_OFF  131072   // attn probs f16 [8][64][64] (65536 B)
#define HB_OFF   196608   // f32 residual / LN buffer [64][256] (65536 B)
#define GB_OFF   262144   // gelu chunk f16, double-buffered 2x[64][32] (8192 B)
#define SMEM_BYTES 270336

#define WMMA(a, b, c) \
  __builtin_amdgcn_wmma_f32_16x16x32_f16(false, (a), false, (b), (short)0, (c), false, false)

// A fragment (16x32 f16, split-K layout per ISA 7.12.2): lanes 0-15 = rows,
// K 0..7 in v0-3 and 16..23 in v4-7; lanes 16-31 shifted by +8 in K.
__device__ __forceinline__ v16h frag_a(const _Float16* base, int ld, int m0, int k0, int lane) {
  const _Float16* p = base + (m0 + (lane & 15)) * ld + k0 + ((lane >> 4) << 3);
  v8h lo = *(const v8h*)(p);
  v8h hi = *(const v8h*)(p + 16);
  return __builtin_shufflevector(lo, hi, 0,1,2,3,4,5,6,7,8,9,10,11,12,13,14,15);
}

// B fragment loaded from B^T storage [N][K] (split-K layout matching frag_a):
// lanes 0-15 hold K=k0..k0+7 and k0+16..k0+23, lanes 16-31 shifted by +8 in K, for column n.
__device__ __forceinline__ v16h frag_b(const _Float16* base, int ld, int n0, int k0, int lane) {
  const _Float16* p = base + (n0 + (lane & 15)) * ld + k0 + ((lane >> 4) << 3);
  v8h lo = *(const v8h*)(p);
  v8h hi = *(const v8h*)(p + 16);
  return __builtin_shufflevector(lo, hi, 0,1,2,3,4,5,6,7,8,9,10,11,12,13,14,15);
}

// LayerNorm: wave w handles rows w*8..w*8+7; writes normalized f32 back to HB
// and f16 copy to X16.
__device__ __forceinline__ void layer_norm_rows(float* HB, _Float16* X16,
                                                const float* gamma, const float* beta,
                                                int w, int lane) {
  for (int rr = 0; rr < 8; ++rr) {
    int row = w * 8 + rr;
    float vals[8];
    float s = 0.f;
#pragma unroll
    for (int j = 0; j < 8; ++j) { vals[j] = HB[row * D_ + lane + 32 * j]; s += vals[j]; }
#pragma unroll
    for (int off = 16; off >= 1; off >>= 1) s += __shfl_xor(s, off);
    float mu = s * (1.0f / D_);
    float vsum = 0.f;
#pragma unroll
    for (int j = 0; j < 8; ++j) { float d = vals[j] - mu; vsum += d * d; }
#pragma unroll
    for (int off = 16; off >= 1; off >>= 1) vsum += __shfl_xor(vsum, off);
    float rstd = rsqrtf(vsum * (1.0f / D_) + 1e-5f);
#pragma unroll
    for (int j = 0; j < 8; ++j) {
      int col = lane + 32 * j;
      float nv = (vals[j] - mu) * rstd * gamma[col] + beta[col];
      HB[row * D_ + col] = nv;
      X16[row * D_ + col] = (_Float16)nv;
    }
  }
}

// Convert + transpose all weights fp32 -> f16 into workspace.
// Layout: [l*4+m][256][256] for Wq/Wk/Wv/Wo (as W^T), then W1^T [l][1024][256],
// then W2^T [l][256][1024].
__global__ __launch_bounds__(256) void prep_weights(const float* __restrict__ Wq,
                                                    const float* __restrict__ Wk,
                                                    const float* __restrict__ Wv,
                                                    const float* __restrict__ Wo,
                                                    const float* __restrict__ W1,
                                                    const float* __restrict__ W2,
                                                    _Float16* __restrict__ ws) {
  int idx = blockIdx.x * 256 + threadIdx.x;
  if (idx < 1048576) {
    int lm = idx >> 16;
    int rem = idx & 65535;
    int n = rem >> 8, k = rem & 255;
    int l = lm >> 2, m = lm & 3;
    const float* W = (m == 0) ? Wq : (m == 1) ? Wk : (m == 2) ? Wv : Wo;
    ws[idx] = (_Float16)W[l * 65536 + k * 256 + n];
  } else if (idx < 2097152) {
    int t = idx - 1048576;
    int l = t >> 18;
    int rem = t & 262143;
    int n = rem >> 8, k = rem & 255;
    ws[idx] = (_Float16)W1[l * 262144 + k * 1024 + n];
  } else if (idx < 3145728) {
    int t = idx - 2097152;
    int l = t >> 18;
    int rem = t & 262143;
    int n = rem >> 10, k = rem & 1023;
    ws[idx] = (_Float16)W2[l * 262144 + k * 256 + n];
  }
}

__global__ __launch_bounds__(256) void deeppm_forward(
    const float* __restrict__ x, const int* __restrict__ maskp, const int* __restrict__ osmp,
    const float* __restrict__ wap,
    const float* __restrict__ bq, const float* __restrict__ bk,
    const float* __restrict__ bv, const float* __restrict__ bo,
    const float* __restrict__ b1, const float* __restrict__ b2,
    const float* __restrict__ g1, const float* __restrict__ be1,
    const float* __restrict__ g2, const float* __restrict__ be2,
    const _Float16* __restrict__ ws, float* __restrict__ out) {
  extern __shared__ char smem[];
  _Float16* XF  = (_Float16*)(smem + XF_OFF);
  _Float16* QB  = (_Float16*)(smem + QB_OFF);
  _Float16* KB  = (_Float16*)(smem + KB_OFF);
  _Float16* VT  = (_Float16*)(smem + VT_OFF);
  _Float16* ATT = (_Float16*)(smem + ATT_OFF);
  float*    HB  = (float*)(smem + HB_OFF);
  _Float16* GB  = (_Float16*)(smem + GB_OFF);
  _Float16* OB  = QB;  // attention output reuses Q buffer (barrier-separated)

  __shared__ int s_mod[S_];
  __shared__ int s_mask[S_];

  const int tid  = threadIdx.x;
  const int lane = tid & 31;
  const int w    = tid >> 5;
  const int seq  = blockIdx.x;

  const int osmv = osmp[seq];
  if (tid < S_) {
    int mv = maskp[seq * S_ + tid];
    s_mask[tid] = mv;
    s_mod[tid]  = (mv != 0) && (osmv == 0);
  }
  for (int i = tid; i < S_ * D_; i += 256) {
    float v = osmv ? 1.0f : x[(size_t)seq * (S_ * D_) + i];
    XF[i] = (_Float16)v;
  }
  __syncthreads();

  for (int l = 0; l < L_; ++l) {
    const _Float16* wqT = ws + (size_t)(l * 4 + 0) * 65536;
    const _Float16* wkT = ws + (size_t)(l * 4 + 1) * 65536;
    const _Float16* wvT = ws + (size_t)(l * 4 + 2) * 65536;
    const _Float16* woT = ws + (size_t)(l * 4 + 3) * 65536;
    const _Float16* w1T = ws + 1048576 + (size_t)l * 262144;
    const _Float16* w2T = ws + 2097152 + (size_t)l * 262144;

    // ---- Q projection: 64 tiles over 8 waves, straight-line stores ----
    for (int t = w; t < 64; t += 8) {
      int mt = t >> 4, nt = t & 15;
      v8f acc = {};
#pragma unroll
      for (int kc = 0; kc < 8; ++kc) {
        v16h a = frag_a(XF, D_, mt * 16, kc * 32, lane);
        v16h b = frag_b(wqT, D_, nt * 16, kc * 32, lane);
        acc = WMMA(a, b, acc);
      }
      int col = nt * 16 + (lane & 15);
      float bval = bq[l * D_ + col];
#pragma unroll
      for (int r = 0; r < 8; ++r) {
        int row = mt * 16 + r + ((lane >> 4) << 3);
        QB[row * D_ + col] = (_Float16)(acc[r] + bval);
      }
    }
    // ---- K projection ----
    for (int t = w; t < 64; t += 8) {
      int mt = t >> 4, nt = t & 15;
      v8f acc = {};
#pragma unroll
      for (int kc = 0; kc < 8; ++kc) {
        v16h a = frag_a(XF, D_, mt * 16, kc * 32, lane);
        v16h b = frag_b(wkT, D_, nt * 16, kc * 32, lane);
        acc = WMMA(a, b, acc);
      }
      int col = nt * 16 + (lane & 15);
      float bval = bk[l * D_ + col];
#pragma unroll
      for (int r = 0; r < 8; ++r) {
        int row = mt * 16 + r + ((lane >> 4) << 3);
        KB[row * D_ + col] = (_Float16)(acc[r] + bval);
      }
    }
    // ---- V projection (stored transposed into VT[d][s]) ----
    for (int t = w; t < 64; t += 8) {
      int mt = t >> 4, nt = t & 15;
      v8f acc = {};
#pragma unroll
      for (int kc = 0; kc < 8; ++kc) {
        v16h a = frag_a(XF, D_, mt * 16, kc * 32, lane);
        v16h b = frag_b(wvT, D_, nt * 16, kc * 32, lane);
        acc = WMMA(a, b, acc);
      }
      int col = nt * 16 + (lane & 15);
      float bval = bv[l * D_ + col];
#pragma unroll
      for (int r = 0; r < 8; ++r) {
        int row = mt * 16 + r + ((lane >> 4) << 3);
        VT[col * S_ + row] = (_Float16)(acc[r] + bval);
      }
    }
    __syncthreads();

    // ---- Attention: one head per wave; softmax in registers ----
    const int h = w;
    const float scale = 0.17677669529663687f;  // 1/sqrt(32)
    for (int mt = 0; mt < 4; ++mt) {
      v8f e0 = {}, e1 = {}, e2 = {}, e3 = {};
      v16h aq = frag_a(QB, D_, mt * 16, h * HD_, lane);
      e0 = WMMA(aq, frag_b(KB, D_, 0,  h * HD_, lane), e0);
      e1 = WMMA(aq, frag_b(KB, D_, 16, h * HD_, lane), e1);
      e2 = WMMA(aq, frag_b(KB, D_, 32, h * HD_, lane), e2);
      e3 = WMMA(aq, frag_b(KB, D_, 48, h * HD_, lane), e3);
      int kcol = lane & 15;
#pragma unroll
      for (int r = 0; r < 8; ++r) {
        int qrow = mt * 16 + r + ((lane >> 4) << 3);
        const float* war = wap + ((size_t)seq * S_ + qrow) * S_;
        bool qm = s_mod[qrow] != 0;
        float v0 = qm ? -1e10f : e0[r] * scale * war[0  + kcol];
        float v1 = qm ? -1e10f : e1[r] * scale * war[16 + kcol];
        float v2 = qm ? -1e10f : e2[r] * scale * war[32 + kcol];
        float v3 = qm ? -1e10f : e3[r] * scale * war[48 + kcol];
        float mx = fmaxf(fmaxf(v0, v1), fmaxf(v2, v3));
#pragma unroll
        for (int off = 8; off >= 1; off >>= 1) mx = fmaxf(mx, __shfl_xor(mx, off));
        v0 = __expf(v0 - mx); v1 = __expf(v1 - mx);
        v2 = __expf(v2 - mx); v3 = __expf(v3 - mx);
        float sm = v0 + v1 + v2 + v3;
#pragma unroll
        for (int off = 8; off >= 1; off >>= 1) sm += __shfl_xor(sm, off);
        float inv = 1.0f / sm;
        _Float16* arow = ATT + h * (S_ * S_) + qrow * S_;
        arow[0  + kcol] = (_Float16)(v0 * inv);
        arow[16 + kcol] = (_Float16)(v1 * inv);
        arow[32 + kcol] = (_Float16)(v2 * inv);
        arow[48 + kcol] = (_Float16)(v3 * inv);
      }
    }
    __syncthreads();

    // ---- attn @ V (per head), write into OB (= QB) ----
    for (int mt = 0; mt < 4; ++mt) {
      v16h a0 = frag_a(ATT + h * (S_ * S_), S_, mt * 16, 0,  lane);
      v16h a1 = frag_a(ATT + h * (S_ * S_), S_, mt * 16, 32, lane);
#pragma unroll
      for (int vn = 0; vn < 2; ++vn) {
        v8f acc = {};
        acc = WMMA(a0, frag_b(VT, S_, h * HD_ + vn * 16, 0,  lane), acc);
        acc = WMMA(a1, frag_b(VT, S_, h * HD_ + vn * 16, 32, lane), acc);
        int col = h * HD_ + vn * 16 + (lane & 15);
#pragma unroll
        for (int r = 0; r < 8; ++r) {
          int row = mt * 16 + r + ((lane >> 4) << 3);
          OB[row * D_ + col] = (_Float16)acc[r];
        }
      }
    }
    __syncthreads();

    // ---- Output projection + residual -> HB (f32) ----
    for (int t = w; t < 64; t += 8) {
      int mt = t >> 4, nt = t & 15;
      v8f acc = {};
#pragma unroll
      for (int kc = 0; kc < 8; ++kc) {
        v16h a = frag_a(OB, D_, mt * 16, kc * 32, lane);
        v16h b = frag_b(woT, D_, nt * 16, kc * 32, lane);
        acc = WMMA(a, b, acc);
      }
      int col = nt * 16 + (lane & 15);
      float bval = bo[l * D_ + col];
#pragma unroll
      for (int r = 0; r < 8; ++r) {
        int row = mt * 16 + r + ((lane >> 4) << 3);
        HB[row * D_ + col] = acc[r] + bval + (float)XF[row * D_ + col];
      }
    }
    __syncthreads();
    layer_norm_rows(HB, XF, g1 + l * D_, be1 + l * D_, w, lane);
    __syncthreads();

    // ---- Fused FFN: gelu(h@W1) chunks of 32 feed K-dim of @W2 ----
    // A-fragments of h (row tile gm) are invariant across chunks: hoist them.
    v8f zero = {};
    v8f acc2[8];
#pragma unroll
    for (int j = 0; j < 8; ++j) acc2[j] = zero;
    const int gm = w & 3, gn = w >> 2;
    v16h a_ff[8];
#pragma unroll
    for (int kc = 0; kc < 8; ++kc) a_ff[kc] = frag_a(XF, D_, gm * 16, kc * 32, lane);

    for (int c = 0; c < 32; ++c) {
      _Float16* GBp = GB + (c & 1) * (S_ * 32);
      v8f ga = {};
#pragma unroll
      for (int kc = 0; kc < 8; ++kc) {
        v16h b = frag_b(w1T, D_, c * 32 + gn * 16, kc * 32, lane);
        ga = WMMA(a_ff[kc], b, ga);
      }
      int dcol = c * 32 + gn * 16 + (lane & 15);
      float bval = b1[l * DFF_ + dcol];
#pragma unroll
      for (int r = 0; r < 8; ++r) {
        int row = gm * 16 + r + ((lane >> 4) << 3);
        float xv = ga[r] + bval;
        float gl = 0.5f * xv * (1.0f + erff(xv * 0.70710678118f));  // exact gelu
        GBp[row * 32 + gn * 16 + (lane & 15)] = (_Float16)gl;
      }
      __syncthreads();  // single barrier per chunk (GB is double-buffered)
#pragma unroll
      for (int mt = 0; mt < 4; ++mt) {
        v16h a = frag_a(GBp, 32, mt * 16, 0, lane);
#pragma unroll
        for (int jj = 0; jj < 2; ++jj) {
          int nt = w * 2 + jj;
          v16h b = frag_b(w2T, DFF_, nt * 16, c * 32, lane);
          acc2[mt * 2 + jj] = WMMA(a, b, acc2[mt * 2 + jj]);
        }
      }
    }
    __syncthreads();
    // finalize FFN: HB += o2 + b2 (HB holds post-LN1 h)
#pragma unroll
    for (int j = 0; j < 8; ++j) {
      int mt = j >> 1;
      int nt = w * 2 + (j & 1);
      int col = nt * 16 + (lane & 15);
      float bval = b2[l * D_ + col];
#pragma unroll
      for (int r = 0; r < 8; ++r) {
        int row = mt * 16 + r + ((lane >> 4) << 3);
        HB[row * D_ + col] += acc2[j][r] + bval;
      }
    }
    __syncthreads();
    layer_norm_rows(HB, XF, g2 + l * D_, be2 + l * D_, w, lane);
    __syncthreads();
  }

  // out = where(mask, 0, xf)  (f32 from HB)
  for (int i = tid; i < S_ * D_; i += 256) {
    int srow = i >> 8;
    out[(size_t)seq * (S_ * D_) + i] = s_mask[srow] ? 0.0f : HB[i];
  }
}

extern "C" void kernel_launch(void* const* d_in, const int* in_sizes, int n_in,
                              void* d_out, int out_size, void* d_ws, size_t ws_size,
                              hipStream_t stream) {
  (void)in_sizes; (void)n_in; (void)out_size; (void)ws_size;
  const float* x    = (const float*)d_in[0];
  const int*  maskp = (const int*)d_in[1];
  const int*  osmp  = (const int*)d_in[2];
  const float* wap  = (const float*)d_in[3];
  const float* Wq   = (const float*)d_in[4];
  const float* bq   = (const float*)d_in[5];
  const float* Wk   = (const float*)d_in[6];
  const float* bk   = (const float*)d_in[7];
  const float* Wv   = (const float*)d_in[8];
  const float* bv   = (const float*)d_in[9];
  const float* Wo   = (const float*)d_in[10];
  const float* bo   = (const float*)d_in[11];
  const float* W1   = (const float*)d_in[12];
  const float* b1   = (const float*)d_in[13];
  const float* W2   = (const float*)d_in[14];
  const float* b2   = (const float*)d_in[15];
  const float* g1   = (const float*)d_in[16];
  const float* be1  = (const float*)d_in[17];
  const float* g2   = (const float*)d_in[18];
  const float* be2  = (const float*)d_in[19];
  _Float16* ws = (_Float16*)d_ws;
  float* out = (float*)d_out;

  prep_weights<<<3145728 / 256, 256, 0, stream>>>(Wq, Wk, Wv, Wo, W1, W2, ws);

  hipFuncSetAttribute((const void*)deeppm_forward,
                      hipFuncAttributeMaxDynamicSharedMemorySize, SMEM_BYTES);
  deeppm_forward<<<NSEQ, 256, SMEM_BYTES, stream>>>(
      x, maskp, osmp, wap, bq, bk, bv, bo, b1, b2, g1, be1, g2, be2, ws, out);
}
